// ImPLE_54537494725030
// MI455X (gfx1250) — matrix operand order repaired and checked by reading, not compiled, not run on hardware
//
#include <hip/hip_runtime.h>
#include <hip/hip_bf16.h>
#include <stdint.h>

// Problem constants (reference: B, T, D, L, E = 4096, 4, 1024, 2, 4)
#define BB 4096
#define TT 4
#define DD 1024
#define LL 2

// GEMM tiling: block 128x64x32, 8 waves, wave tile 32x32 (2x2 WMMA tiles)
#define BM 128
#define BN 64
#define BK 32

typedef __attribute__((ext_vector_type(16))) __bf16 v16bf;
typedef __attribute__((ext_vector_type(8)))  float  v8f;

struct FragBF {
    union { uint4 q[2]; v16bf v; };
};

union Pack2 { __bf16 h[2]; uint32_t u; };

__device__ __forceinline__ __bf16 f2bf(float x) { return (__bf16)x; }

__device__ __forceinline__ uint32_t pack_bf2(__bf16 a, __bf16 b) {
    Pack2 p; p.h[0] = a; p.h[1] = b; return p.u;
}

// C[b, t, n] = relu( sum_d A[b, t, d] * W[l, t, d, n] + bias[l, t, n] ) * scale
// scale = sum_e inte[l,t,e] if inte != nullptr, else 1.
// Split-precision bf16x3: fp32 -> hi(bf16) + lo(bf16 residual); 3 WMMAs/tile/K-step.
__global__ __launch_bounds__(256) void imple_gemm_bf16x3(
    const float* __restrict__ A,
    const float* __restrict__ Wall,
    const float* __restrict__ ball,
    const float* __restrict__ inteall,   // may be nullptr
    float* __restrict__ C,
    int layer)
{
    const int t    = blockIdx.z;
    const int m0   = blockIdx.x * BM;
    const int n0   = blockIdx.y * BN;
    const int tid  = threadIdx.x;
    const int lane = tid & 31;
    const int wave = tid >> 5;
    const int waveM = wave & 3;   // 4 M-strips of 32 rows
    const int waveN = wave >> 2;  // 2 N-strips of 32 cols

    const float* __restrict__ Wt   = Wall + ((size_t)(layer * TT + t)) * DD * DD;
    const float* __restrict__ bias = ball + (size_t)(layer * TT + t) * DD;

    float s = 1.0f;
    if (inteall) {
        const float* ip = inteall + (size_t)(layer * TT + t) * 4;
        s = ip[0] + ip[1] + ip[2] + ip[3];
    }

    // LDS: A tile [BM][BK] (k contiguous), W tile transposed [BN][BK]
    __shared__ __align__(16) __bf16 AsHi[BM * BK];
    __shared__ __align__(16) __bf16 AsLo[BM * BK];
    __shared__ __align__(16) __bf16 WsHi[BN * BK];
    __shared__ __align__(16) __bf16 WsLo[BN * BK];

    v8f acc[2][2] = {{{}, {}}, {{}, {}}};

    const int hl   = lane >> 4;   // half-wave select
    const int nloc = lane & 15;

    // staging coordinates (A): 128 rows x 32 cols, 4 passes of float4
    const int ar = tid >> 3;          // 0..31 (row base, +32 per pass)
    const int ac = (tid & 7) * 4;     // 0..28
    // staging coordinates (W): thread owns n-column, 8 consecutive k
    const int wn = tid & 63;          // 0..63
    const int wk = (tid >> 6) * 8;    // 0, 8, 16, 24

    for (int kk = 0; kk < DD; kk += BK) {
        // ---- prefetch next K-tile into L2 (global_prefetch_b8) ----
        if (kk + BK < DD) {
            __builtin_prefetch(A + ((size_t)(m0 + ar) * TT + t) * DD + kk + BK + ac, 0, 1);
            __builtin_prefetch(Wt + (size_t)(kk + BK + wk) * DD + n0 + wn, 0, 1);
        }

        // ---- stage A: fp32 -> bf16 hi/lo, packed 8-byte LDS stores ----
        #pragma unroll
        for (int p = 0; p < 4; ++p) {
            const int r = ar + p * 32;
            const float4 av = *(const float4*)(A + ((size_t)(m0 + r) * TT + t) * DD + kk + ac);
            const float af[4] = {av.x, av.y, av.z, av.w};
            __bf16 hi[4], lo[4];
            #pragma unroll
            for (int i = 0; i < 4; ++i) {
                hi[i] = f2bf(af[i]);
                lo[i] = f2bf(af[i] - (float)hi[i]);
            }
            uint2 h2, l2;
            h2.x = pack_bf2(hi[0], hi[1]); h2.y = pack_bf2(hi[2], hi[3]);
            l2.x = pack_bf2(lo[0], lo[1]); l2.y = pack_bf2(lo[2], lo[3]);
            *(uint2*)&AsHi[r * BK + ac] = h2;
            *(uint2*)&AsLo[r * BK + ac] = l2;
        }

        // ---- stage W transposed: 8 coalesced b32 loads, one b128 store ----
        {
            float wf[8];
            #pragma unroll
            for (int i = 0; i < 8; ++i)
                wf[i] = Wt[(size_t)(kk + wk + i) * DD + n0 + wn];
            __bf16 hi[8], lo[8];
            #pragma unroll
            for (int i = 0; i < 8; ++i) {
                hi[i] = f2bf(wf[i]);
                lo[i] = f2bf(wf[i] - (float)hi[i]);
            }
            uint4 h4, l4;
            h4.x = pack_bf2(hi[0], hi[1]); h4.y = pack_bf2(hi[2], hi[3]);
            h4.z = pack_bf2(hi[4], hi[5]); h4.w = pack_bf2(hi[6], hi[7]);
            l4.x = pack_bf2(lo[0], lo[1]); l4.y = pack_bf2(lo[2], lo[3]);
            l4.z = pack_bf2(lo[4], lo[5]); l4.w = pack_bf2(lo[6], lo[7]);
            *(uint4*)&WsHi[wn * BK + wk] = h4;
            *(uint4*)&WsLo[wn * BK + wk] = l4;
        }
        __syncthreads();

        // ---- fragments (ISA 7.12.2 16-bit layouts) + 12 WMMAs per wave ----
        const uint4* AhQ = (const uint4*)AsHi;  // 4 uint4 per 32-bf16 row
        const uint4* AlQ = (const uint4*)AsLo;
        const uint4* WhQ = (const uint4*)WsHi;
        const uint4* WlQ = (const uint4*)WsLo;

        FragBF ah[2], al[2], bh[2], bl[2];
        #pragma unroll
        for (int mi = 0; mi < 2; ++mi) {
            const int m = 32 * waveM + 16 * mi + nloc;   // A row for this lane
            ah[mi].q[0] = AhQ[m * 4 + hl];     ah[mi].q[1] = AhQ[m * 4 + 2 + hl];
            al[mi].q[0] = AlQ[m * 4 + hl];     al[mi].q[1] = AlQ[m * 4 + 2 + hl];
        }
        #pragma unroll
        for (int nj = 0; nj < 2; ++nj) {
            const int cn = 32 * waveN + 16 * nj + nloc;  // B column for this lane
            bh[nj].q[0] = WhQ[cn * 4 + hl]; bh[nj].q[1] = WhQ[cn * 4 + 2 + hl];
            bl[nj].q[0] = WlQ[cn * 4 + hl]; bl[nj].q[1] = WlQ[cn * 4 + 2 + hl];
        }

        #pragma unroll
        for (int mi = 0; mi < 2; ++mi) {
            #pragma unroll
            for (int nj = 0; nj < 2; ++nj) {
                v8f a = acc[mi][nj];
                a = __builtin_amdgcn_wmma_f32_16x16x32_bf16(false, ah[mi].v, false, bh[nj].v,
                                                            (short)0, a, false, false);
                a = __builtin_amdgcn_wmma_f32_16x16x32_bf16(false, al[mi].v, false, bh[nj].v,
                                                            (short)0, a, false, false);
                a = __builtin_amdgcn_wmma_f32_16x16x32_bf16(false, ah[mi].v, false, bl[nj].v,
                                                            (short)0, a, false, false);
                acc[mi][nj] = a;
            }
        }
        __syncthreads();
    }

    // ---- epilogue: bias + ReLU (+scale), scatter per C-tile layout ----
    #pragma unroll
    for (int nj = 0; nj < 2; ++nj) {
        const int col = n0 + 32 * waveN + 16 * nj + nloc;
        const float bj = bias[col];
        #pragma unroll
        for (int mi = 0; mi < 2; ++mi) {
            #pragma unroll
            for (int r = 0; r < 8; ++r) {
                const int row = m0 + 32 * waveM + 16 * mi + r + 8 * hl;
                float v = acc[mi][nj][r] + bj;
                v = fmaxf(v, 0.0f) * s;
                C[((size_t)row * TT + t) * DD + col] = v;
            }
        }
    }
}

// g[row, k] = softmax_k( sum_d teo[row, d] * Wg[l, d, k] + bg[l, k] ), row = b*T + t
__global__ __launch_bounds__(128) void imple_gate(
    const float* __restrict__ teo,
    const float* __restrict__ Wg,
    const float* __restrict__ bg,
    float* __restrict__ g,
    int layer)
{
    const int row = blockIdx.x;
    const int tid = threadIdx.x;
    const float* tr = teo + (size_t)row * DD;
    const float* wg = Wg + (size_t)layer * DD * TT;

    float p[4] = {0.f, 0.f, 0.f, 0.f};
    for (int d = tid; d < DD; d += 128) {
        const float v = tr[d];
        #pragma unroll
        for (int k = 0; k < 4; ++k) p[k] += v * wg[d * TT + k];
    }

    __shared__ float red[128 * 4];
    #pragma unroll
    for (int k = 0; k < 4; ++k) red[tid * 4 + k] = p[k];
    __syncthreads();
    for (int st = 64; st > 0; st >>= 1) {
        if (tid < st) {
            #pragma unroll
            for (int k = 0; k < 4; ++k) red[tid * 4 + k] += red[(tid + st) * 4 + k];
        }
        __syncthreads();
    }
    if (tid == 0) {
        float v[4];
        #pragma unroll
        for (int k = 0; k < 4; ++k) v[k] = red[k] + bg[layer * TT + k];
        float mx = fmaxf(fmaxf(v[0], v[1]), fmaxf(v[2], v[3]));
        float e[4], sum = 0.f;
        #pragma unroll
        for (int k = 0; k < 4; ++k) { e[k] = __expf(v[k] - mx); sum += e[k]; }
        const float inv = 1.0f / sum;
        #pragma unroll
        for (int k = 0; k < 4; ++k) g[(size_t)row * 4 + k] = e[k] * inv;
    }
}

// h[b, t, d] = sum_k g[b, t, k] * teo[b, k, d] + teo[b, t, d]
// Safe in-place (hout == teo): each thread reads its 4 task values before
// writing, and each (b, d) pair is owned by exactly one thread.
__global__ __launch_bounds__(256) void imple_combine(
    const float* teo, const float* g, float* hout)
{
    const int b = blockIdx.x;
    const int d = blockIdx.y * 256 + threadIdx.x;
    const float* gb = g + (size_t)b * 16;
    float gm[16];
    #pragma unroll
    for (int i = 0; i < 16; ++i) gm[i] = gb[i];
    float te[4];
    #pragma unroll
    for (int k = 0; k < 4; ++k) te[k] = teo[((size_t)b * TT + k) * DD + d];
    float o[4];
    #pragma unroll
    for (int t = 0; t < 4; ++t) {
        float v = te[t];
        #pragma unroll
        for (int k = 0; k < 4; ++k) v += gm[t * 4 + k] * te[k];
        o[t] = v;
    }
    #pragma unroll
    for (int t = 0; t < 4; ++t) hout[((size_t)b * TT + t) * DD + d] = o[t];
}

extern "C" void kernel_launch(void* const* d_in, const int* in_sizes, int n_in,
                              void* d_out, int out_size, void* d_ws, size_t ws_size,
                              hipStream_t stream) {
    const float* x    = (const float*)d_in[0];
    const float* W1   = (const float*)d_in[1];
    const float* b1   = (const float*)d_in[2];
    const float* W2   = (const float*)d_in[3];
    const float* b2   = (const float*)d_in[4];
    const float* inte = (const float*)d_in[5];
    const float* Wg   = (const float*)d_in[6];
    const float* bg   = (const float*)d_in[7];
    float* out = (float*)d_out;   // doubles as the teo buffer each layer

    const size_t nBTD = (size_t)BB * TT * DD;
    float* t1   = (float*)d_ws;          // [B,T,D] scratch (t1)
    float* hbuf = t1 + nBTD;             // [B,T,D] scratch (inter-layer h)
    float* gbuf = hbuf + nBTD;           // [B,T,T] gate weights

    const dim3 ggrid(BB / BM, DD / BN, TT);   // 32 x 16 x 4 blocks
    const dim3 gateg(BB * TT);
    const dim3 combg(BB, DD / 256);

    for (int l = 0; l < LL; ++l) {
        const float* hin = (l == 0) ? x : hbuf;
        // t1 = relu(h @ W1 + b1)
        imple_gemm_bf16x3<<<ggrid, 256, 0, stream>>>(hin, W1, b1, nullptr, t1, l);
        // teo = relu(t1 @ W2 + b2) * sum_e(inte)   (teo lives in d_out)
        imple_gemm_bf16x3<<<ggrid, 256, 0, stream>>>(t1, W2, b2, inte, out, l);
        // g = softmax(teo @ Wg + bg)
        imple_gate<<<gateg, 128, 0, stream>>>(out, Wg, bg, gbuf, l);
        // h = g @ teo + teo  (final layer writes d_out in-place)
        float* hdst = (l == 0) ? hbuf : out;
        imple_combine<<<combg, 256, 0, stream>>>(out, gbuf, hdst);
    }
}
